// Spectre_88596585382378
// MI455X (gfx1250) — matrix operand order repaired and checked by reading, not compiled, not run on hardware
//
#include <hip/hip_runtime.h>
#include <hip/hip_bf16.h>
#include <math.h>

// Problem constants (from reference)
#define BB_  4
#define NN_  8192
#define DD_  1024
#define HH_  16
#define DHH_ 64
#define GHH_ 256
#define INV_SQRT2_ 0.70710678118654752440f

typedef __attribute__((ext_vector_type(16))) __bf16 v16bf;
typedef __attribute__((ext_vector_type(8)))  float v8f;

__device__ __forceinline__ unsigned short f32_to_bf16(float f) {
  union { float f; unsigned int u; } x; x.f = f;
  unsigned int r = x.u + 0x7FFFu + ((x.u >> 16) & 1u);
  return (unsigned short)(r >> 16);
}

__device__ __forceinline__ float gelu_exact(float x) {
  return 0.5f * x * (1.0f + erff(x * INV_SQRT2_));
}

// gfx1250 async global->LDS copy of 16 bytes (tracked by ASYNCcnt).
__device__ __forceinline__ void async_g2l_b128(unsigned lds_off,
                                               const void* gaddr) {
  asm volatile("global_load_async_to_lds_b128 %0, %1, off"
               :: "v"(lds_off), "v"(gaddr)
               : "memory");
}
__device__ __forceinline__ void wait_asynccnt0() {
  asm volatile("s_wait_asynccnt 0x0" ::: "memory");
}

// ---------------------------------------------------------------------------
// Kernel 1: fused  xbar[b,d] = mean_n x[b,n,d]  and  xbf = bf16(x)
// grid (D/256, B), block 256. Coalesced along d.
// ---------------------------------------------------------------------------
__global__ __launch_bounds__(256) void k_xbar_xbf(
    const float* __restrict__ x, float* __restrict__ xbar,
    unsigned short* __restrict__ xbf) {
  const int d = blockIdx.x * 256 + threadIdx.x;
  const int b = blockIdx.y;
  const float* xp = x + (size_t)b * NN_ * DD_ + d;
  unsigned short* yp = xbf + (size_t)b * NN_ * DD_ + d;
  float acc = 0.0f;
  for (int n = 0; n < NN_; ++n) {
    float v = xp[(size_t)n * DD_];
    acc += v;
    yp[(size_t)n * DD_] = f32_to_bf16(v);
  }
  xbar[b * DD_ + d] = acc * (1.0f / (float)NN_);
}

// ---------------------------------------------------------------------------
// Kernel 2: elementwise f32 -> bf16 (weights)
// ---------------------------------------------------------------------------
__global__ __launch_bounds__(256) void k_cvt_bf16(
    const float* __restrict__ src, unsigned short* __restrict__ dst, int n) {
  int i = blockIdx.x * 256 + threadIdx.x;
  if (i < n) dst[i] = f32_to_bf16(src[i]);
}

// ---------------------------------------------------------------------------
// Kernel 3: per-(b,h) head: bar_q = xbar@w_q col block, LayerNorm over DH,
// gelu hidden activations for gate and wrm MLPs. grid = B*H, block 256.
// hidden layout: [bh][2][GH]
// ---------------------------------------------------------------------------
__global__ __launch_bounds__(256) void k_head(
    const float* __restrict__ xbar, const float* __restrict__ w_q,
    const float* __restrict__ ln_g, const float* __restrict__ ln_b,
    const float* __restrict__ gate_w1, const float* __restrict__ wrm_w1,
    float* __restrict__ hidden) {
  const int bh = blockIdx.x;
  const int b = bh / HH_, h = bh % HH_;
  const int t = threadIdx.x;
  __shared__ float bq[DHH_];

  if (t < DHH_) {
    const float* wq = w_q + (h * DHH_ + t);   // column (stride D)
    const float* xb = xbar + b * DD_;
    float acc = 0.0f;
    for (int k = 0; k < DD_; ++k) acc += xb[k] * wq[(size_t)k * DD_];
    bq[t] = acc;
  }
  __syncthreads();
  if (t == 0) {
    float mu = 0.0f;
    for (int i = 0; i < DHH_; ++i) mu += bq[i];
    mu *= (1.0f / DHH_);
    float var = 0.0f;
    for (int i = 0; i < DHH_; ++i) { float dlt = bq[i] - mu; var += dlt * dlt; }
    var *= (1.0f / DHH_);
    float inv = rsqrtf(var + 1e-5f);
    for (int i = 0; i < DHH_; ++i) bq[i] = (bq[i] - mu) * inv * ln_g[i] + ln_b[i];
  }
  __syncthreads();
  // t indexes hidden unit j in [0,256)
  float a1 = 0.0f, a2 = 0.0f;
  for (int k = 0; k < DHH_; ++k) {
    float q = bq[k];
    a1 += q * gate_w1[k * GHH_ + t];
    a2 += q * wrm_w1[k * GHH_ + t];
  }
  hidden[((size_t)bh * 2 + 0) * GHH_ + t] = gelu_exact(a1);
  hidden[((size_t)bh * 2 + 1) * GHH_ + t] = gelu_exact(a2);
}

// ---------------------------------------------------------------------------
// Kernel 4: filters: g_re/g_im (complex spectral gate) and s (write-mix scale)
// grid (N/256, B*H), block 256. Coalesced over n.
// ---------------------------------------------------------------------------
__global__ __launch_bounds__(256) void k_filters(
    const float* __restrict__ hidden, const float* __restrict__ gate_w2,
    const float* __restrict__ wrm_w2, float* __restrict__ gre,
    float* __restrict__ gim, float* __restrict__ sbuf) {
  const int bh = blockIdx.y;
  const int n = blockIdx.x * 256 + threadIdx.x;
  const int t = threadIdx.x;
  __shared__ float hg[GHH_], hw[GHH_];
  hg[t] = hidden[((size_t)bh * 2 + 0) * GHH_ + t];
  hw[t] = hidden[((size_t)bh * 2 + 1) * GHH_ + t];
  __syncthreads();
  float are = 0.0f, aim = 0.0f, as = 0.0f;
  for (int j = 0; j < GHH_; ++j) {
    float hgj = hg[j], hwj = hw[j];
    are += hgj * gate_w2[(size_t)j * (2 * NN_) + n];
    aim += hgj * gate_w2[(size_t)j * (2 * NN_) + NN_ + n];
    as  += hwj * wrm_w2[(size_t)j * NN_ + n];
  }
  size_t o = (size_t)bh * NN_ + n;
  gre[o] = are; gim[o] = aim; sbuf[o] = as;
}

// ---------------------------------------------------------------------------
// Kernel 5/7: bf16 WMMA GEMM, C(f32) = A(bf16, MxK) * B(bf16, KxN)
//  AMODE 0: A row-major [m*K + k]; staged via global_load_async_to_lds_b128
//  AMODE 1: A is (B, D, N) bf16, element (m,k) = A[(b*D + k)*N + n]  (A = y_t)
//  OMODE 0: C[m*D + n]                  (final output, (B,N,D))
//  OMODE 1: C[(b*D + n)*N + nn]         (v in (B, D, N) layout for FFT)
// Tile: 64x128, K-step 32. 8 waves, each wave 2x2 frags of 16x16.
// B tile staged TRANSPOSED in LDS ([col][k]) so fragment reads are 2x b128.
// ---------------------------------------------------------------------------
#define GEMM_BM 64
#define GEMM_BN 128
#define GEMM_BK 32
#define LDA_PAD  40   // ushorts per sA row  (80B, 16B aligned)
#define LDBT_PAD 40   // ushorts per sBt row (80B, 16B aligned)

template <int AMODE, int OMODE>
__global__ __launch_bounds__(256) void k_gemm_bf16(
    const unsigned short* __restrict__ A, const unsigned short* __restrict__ Bw,
    float* __restrict__ C) {
  __shared__ __align__(16) unsigned short sA[GEMM_BM * LDA_PAD];     // 5.0 KB
  __shared__ __align__(16) unsigned short sBt[GEMM_BN * LDBT_PAD];   // 10.0 KB
  const int t = threadIdx.x;
  const int lane = t & 31;
  const int wave = t >> 5;
  const int wm = wave >> 2;   // 0..1
  const int wn = wave & 3;    // 0..3
  const int m0 = blockIdx.y * GEMM_BM;
  const int c0 = blockIdx.x * GEMM_BN;
  const int bA = m0 / NN_;    // batch (AMODE 1)
  const int n0A = m0 % NN_;

  union Frag { v16bf v; unsigned int u[8]; uint4 u4[2]; };
  v8f acc[2][2];
#pragma unroll
  for (int i = 0; i < 2; ++i)
#pragma unroll
    for (int j = 0; j < 2; ++j) acc[i][j] = {};

  for (int k0 = 0; k0 < DD_; k0 += GEMM_BK) {
    // ---- stage A tile into LDS ----
    if (AMODE == 0) {
      // direct row-major copy: gfx1250 async global->LDS, no VGPR round-trip
      const int row = t >> 2, seg = t & 3;           // 64 rows x 4 x b128
      const void* gsrc = (const void*)(A + (size_t)(m0 + row) * DD_ + k0 + seg * 8);
      const unsigned ldst = (unsigned)(size_t)(sA + row * LDA_PAD + seg * 8);
      async_g2l_b128(ldst, gsrc);
    } else {
      // transposed read of y_t (B, D, N): gather along n, scatter to [m][k]
      const int kk = t >> 3, seg = t & 7;            // 32 k x 8 x b128 (along n)
      union { uint4 q; unsigned short s[8]; } u;
      u.q = *(const uint4*)(A + (size_t)(bA * DD_ + k0 + kk) * NN_ + n0A + seg * 8);
#pragma unroll
      for (int i = 0; i < 8; ++i) sA[(seg * 8 + i) * LDA_PAD + kk] = u.s[i];
    }
    // ---- stage B tile TRANSPOSED into LDS: sBt[col][k] ----
#pragma unroll
    for (int it = 0; it < 2; ++it) {
      const int idx = t + it * 256;
      const int kk = idx >> 4, seg = idx & 15;
      union { uint4 q; unsigned short s[8]; } u;
      u.q = *(const uint4*)(Bw + (size_t)(k0 + kk) * DD_ + c0 + seg * 8);
#pragma unroll
      for (int i = 0; i < 8; ++i) sBt[(seg * 8 + i) * LDBT_PAD + kk] = u.s[i];
    }
    if (k0 + GEMM_BK < DD_) {  // gfx1250 global_prefetch path for next tiles
      __builtin_prefetch(Bw + (size_t)(k0 + GEMM_BK) * DD_ + c0, 0, 1);
      if (AMODE == 0)
        __builtin_prefetch(A + (size_t)m0 * DD_ + k0 + GEMM_BK, 0, 1);
    }
    if (AMODE == 0) wait_asynccnt0();   // this wave's async copies landed
    __syncthreads();                    // all waves' staging visible

    // ---- build fragments (A reads are b32, B reads are b128 from LDS) ----
    Frag afr[2], bfr[2];
    {
      const int r = lane & 15, hh = lane >> 4;
#pragma unroll
      for (int fm = 0; fm < 2; ++fm) {
        const unsigned short* rowp = sA + (wm * 32 + fm * 16 + r) * LDA_PAD;
#pragma unroll
        for (int j = 0; j < 8; ++j) {
          const int kk = (j < 4) ? (hh * 8 + 2 * j) : (16 + hh * 8 + 2 * (j - 4));
          afr[fm].u[j] = *(const unsigned int*)(rowp + kk);
        }
      }
#pragma unroll
      for (int fn = 0; fn < 2; ++fn) {
        const unsigned short* cp =
            sBt + (wn * 32 + fn * 16 + r) * LDBT_PAD + hh * 8;
        bfr[fn].u4[0] = *(const uint4*)(cp);      // k = hh*8 + 0..7
        bfr[fn].u4[1] = *(const uint4*)(cp + 16); // k = 16 + hh*8 + 0..7
      }
    }
#pragma unroll
    for (int fm = 0; fm < 2; ++fm)
#pragma unroll
      for (int fn = 0; fn < 2; ++fn)
        acc[fm][fn] = __builtin_amdgcn_wmma_f32_16x16x32_bf16(
            false, afr[fm].v, false, bfr[fn].v, (short)0, acc[fm][fn],
            false, false);
    __syncthreads();
  }

  // ---- epilogue: C/D frag layout: VGPR i -> M = i + 8*(lane/16), N = lane%16
  const int r = lane & 15, hh = lane >> 4;
#pragma unroll
  for (int fm = 0; fm < 2; ++fm) {
#pragma unroll
    for (int fn = 0; fn < 2; ++fn) {
#pragma unroll
      for (int i = 0; i < 8; ++i) {
        const int m = m0 + wm * 32 + fm * 16 + hh * 8 + i;
        const int n = c0 + wn * 32 + fn * 16 + r;
        const float vv = acc[fm][fn][i];
        if (OMODE == 0) {
          C[(size_t)m * DD_ + n] = vv;
        } else {
          const int b = m / NN_, nn = m % NN_;
          C[(size_t)(b * DD_ + n) * NN_ + nn] = vv;
        }
      }
    }
  }
}

// ---------------------------------------------------------------------------
// Kernel 6: spectral conv + Hadamard mixes. One block per (b, d=h*DH+dh)
// column; 8192-pt complex FFT in 64KB LDS; forward FFT, *g, inverse via
// conj-FFT, then even/odd mix * s, low/high mix, residual add, bf16 store.
// ---------------------------------------------------------------------------
__device__ __forceinline__ void fft8192_lds(float* re, float* im, int t) {
  // bit reversal (13 bits)
  for (int n = t; n < NN_; n += 256) {
    const int rv = (int)(__brev((unsigned)n) >> 19);
    if (n < rv) {
      float a = re[n]; re[n] = re[rv]; re[rv] = a;
      float c = im[n]; im[n] = im[rv]; im[rv] = c;
    }
  }
  __syncthreads();
  for (int lg = 1; lg <= 13; ++lg) {
    const int half = 1 << (lg - 1);
    const float ang0 = -6.28318530717958647692f / (float)(1 << lg);
    for (int p = t; p < (NN_ >> 1); p += 256) {
      const int j = p & (half - 1);
      const int i0 = ((p >> (lg - 1)) << lg) + j;
      const int i1 = i0 + half;
      float s, c;
      __sincosf(ang0 * (float)j, &s, &c);
      const float xr = re[i1], xi = im[i1];
      const float tr = xr * c - xi * s;
      const float ti = xr * s + xi * c;
      re[i1] = re[i0] - tr; im[i1] = im[i0] - ti;
      re[i0] += tr;         im[i0] += ti;
    }
    __syncthreads();
  }
}

__global__ __launch_bounds__(256) void k_fftconv(
    const float* __restrict__ v_ws, const float* __restrict__ gre,
    const float* __restrict__ gim, const float* __restrict__ sbuf,
    unsigned short* __restrict__ y_t) {
  __shared__ float re[NN_];
  __shared__ float im[NN_];
  const int col = blockIdx.x;           // b*D + (h*DH + dh)
  const int b = col / DD_, d = col % DD_, h = d / DHH_;
  const int bh = b * HH_ + h;
  const int t = threadIdx.x;
  const float* vp = v_ws + (size_t)col * NN_;
  const float* gr = gre + (size_t)bh * NN_;
  const float* gi = gim + (size_t)bh * NN_;
  const float* sp = sbuf + (size_t)bh * NN_;

  for (int n = t; n < NN_; n += 256) { re[n] = vp[n]; im[n] = 0.0f; }
  __syncthreads();
  fft8192_lds(re, im, t);               // V_hat = FFT(v)
  // multiply by g, then conjugate (inverse = conj(FFT(conj(X)))/N; we only
  // need the real part, so the outer conj is dropped)
  for (int n = t; n < NN_; n += 256) {
    const float a = re[n], c = im[n];
    const float grv = gr[n], giv = gi[n];
    re[n] = a * grv - c * giv;
    im[n] = -(a * giv + c * grv);
  }
  __syncthreads();
  fft8192_lds(re, im, t);
  // v_tilde = real part / N  -> stash in im[]
  for (int n = t; n < NN_; n += 256) im[n] = re[n] * (1.0f / (float)NN_);
  __syncthreads();
  // w_hat = concat((e+o), (e-o)) * inv_sqrt2 * s  -> into re[]
  for (int i = t; i < (NN_ >> 1); i += 256) {
    const float e = im[2 * i], o = im[2 * i + 1];
    re[i]              = (e + o) * INV_SQRT2_ * sp[i];
    re[i + (NN_ >> 1)] = (e - o) * INV_SQRT2_ * sp[i + (NN_ >> 1)];
  }
  __syncthreads();
  // v_ref interleave of ((lo+hi), (lo-hi)) * inv_sqrt2; out = v_tilde + v_ref
  unsigned short* yp = y_t + (size_t)col * NN_;
  for (int i = t; i < (NN_ >> 1); i += 256) {
    const float lo = re[i], hi = re[i + (NN_ >> 1)];
    const float ev = (lo + hi) * INV_SQRT2_;
    const float od = (lo - hi) * INV_SQRT2_;
    yp[2 * i]     = f32_to_bf16(im[2 * i] + ev);
    yp[2 * i + 1] = f32_to_bf16(im[2 * i + 1] + od);
  }
}

// ---------------------------------------------------------------------------
extern "C" void kernel_launch(void* const* d_in, const int* in_sizes, int n_in,
                              void* d_out, int out_size, void* d_ws,
                              size_t ws_size, hipStream_t stream) {
  const float* x       = (const float*)d_in[0];
  const float* w_q     = (const float*)d_in[1];
  const float* w_v     = (const float*)d_in[2];
  const float* w_o     = (const float*)d_in[3];
  const float* ln_g    = (const float*)d_in[4];
  const float* ln_b    = (const float*)d_in[5];
  const float* gate_w1 = (const float*)d_in[6];
  const float* gate_w2 = (const float*)d_in[7];
  const float* wrm_w1  = (const float*)d_in[8];
  const float* wrm_w2  = (const float*)d_in[9];
  float* out = (float*)d_out;

  char* ws = (char*)d_ws;
  size_t off = 0;
  auto alloc = [&](size_t bytes) -> char* {
    off = (off + 255) & ~(size_t)255;
    char* p = ws + off;
    off += bytes;
    return p;
  };
  unsigned short* xbf   = (unsigned short*)alloc((size_t)BB_ * NN_ * DD_ * 2);
  unsigned short* wv_bf = (unsigned short*)alloc((size_t)DD_ * DD_ * 2);
  unsigned short* wo_bf = (unsigned short*)alloc((size_t)DD_ * DD_ * 2);
  float* xbar   = (float*)alloc((size_t)BB_ * DD_ * 4);
  float* hidden = (float*)alloc((size_t)BB_ * HH_ * 2 * GHH_ * 4);
  float* gre    = (float*)alloc((size_t)BB_ * HH_ * NN_ * 4);
  float* gim    = (float*)alloc((size_t)BB_ * HH_ * NN_ * 4);
  float* sbuf   = (float*)alloc((size_t)BB_ * HH_ * NN_ * 4);
  float* v_ws   = (float*)alloc((size_t)BB_ * DD_ * NN_ * 4);
  unsigned short* y_t = (unsigned short*)alloc((size_t)BB_ * DD_ * NN_ * 2);
  (void)ws_size; (void)in_sizes; (void)n_in; (void)out_size;

  // 1) xbar + bf16(x)
  k_xbar_xbf<<<dim3(DD_ / 256, BB_), 256, 0, stream>>>(x, xbar, xbf);
  // 2) weight converts
  k_cvt_bf16<<<(DD_ * DD_) / 256, 256, 0, stream>>>(w_v, wv_bf, DD_ * DD_);
  k_cvt_bf16<<<(DD_ * DD_) / 256, 256, 0, stream>>>(w_o, wo_bf, DD_ * DD_);
  // 3) head (bar_q = xbar @ w_q, LN, gelu hiddens)
  k_head<<<BB_ * HH_, 256, 0, stream>>>(xbar, w_q, ln_g, ln_b, gate_w1,
                                        wrm_w1, hidden);
  // 4) spectral gate + scale
  k_filters<<<dim3(NN_ / 256, BB_ * HH_), 256, 0, stream>>>(
      hidden, gate_w2, wrm_w2, gre, gim, sbuf);
  // 5) v = x @ w_v  (WMMA bf16, output in (B, D, N) layout)
  k_gemm_bf16<0, 1><<<dim3(DD_ / GEMM_BN, (BB_ * NN_) / GEMM_BM), 256, 0,
                      stream>>>(xbf, wv_bf, v_ws);
  // 6) FFT conv + Hadamard mixes + residual -> y bf16 (B, D, N)
  k_fftconv<<<BB_ * DD_, 256, 0, stream>>>(v_ws, gre, gim, sbuf, y_t);
  // 7) out = y @ w_o  (WMMA bf16, transposed A reads, output (B, N, D))
  k_gemm_bf16<1, 0><<<dim3(DD_ / GEMM_BN, (BB_ * NN_) / GEMM_BM), 256, 0,
                      stream>>>(y_t, wo_bf, out);
}
